// Decoder_12867722019365
// MI455X (gfx1250) — matrix operand order repaired and checked by reading, not compiled, not run on hardware
//
#include <hip/hip_runtime.h>
#include <hip/hip_bf16.h>

typedef __attribute__((ext_vector_type(16))) _Float16 v16h;
typedef __attribute__((ext_vector_type(8)))  float    v8f;

#define DEV __device__ __forceinline__

#ifndef USE_ASYNC
#define USE_ASYNC 1   // global_load_async_to_lds_b128 staging of h_EV
#endif

constexpr int B_   = 2;
constexpr int N_   = 2048;
constexpr int H_   = 128;
constexpr int K_   = 30;
constexpr int KP   = 32;     // padded edge rows for WMMA
constexpr int EPAD = 64;     // padded edge-feature dim (39 -> 64)
constexpr int E_IN = 39;
constexpr int NRBF = 16;
constexpr int SEQN_ = 30;
constexpr int DEPTH_ = 3;
constexpr int NNODE = B_ * N_;
constexpr int NEDGE = NNODE * K_;

constexpr size_t alignup(size_t x) { return (x + 255) & ~(size_t)255; }

// workspace layout (bytes)
constexpr size_t OFF_O    = 0;                                                   // f32 [NNODE*9]
constexpr size_t OFF_V    = alignup(OFF_O    + sizeof(float)   * NNODE * 9);     // f32 [NNODE*6]
constexpr size_t OFF_EIDX = alignup(OFF_V    + sizeof(float)   * NNODE * 6);     // i32 [NEDGE]
constexpr size_t OFF_VM   = alignup(OFF_EIDX + sizeof(int)     * NEDGE);         // f32 [NEDGE]
constexpr size_t OFF_EF   = alignup(OFF_VM   + sizeof(float)   * NEDGE);         // f16 [NEDGE*EPAD]
constexpr size_t OFF_HA   = alignup(OFF_EF   + sizeof(_Float16)* (size_t)NEDGE * EPAD); // f32 [NNODE*H]
constexpr size_t OFF_HB   = alignup(OFF_HA   + sizeof(float)   * NNODE * H_);
constexpr size_t OFF_H16A = alignup(OFF_HB   + sizeof(float)   * NNODE * H_);    // f16 [NNODE*H]
constexpr size_t OFF_H16B = alignup(OFF_H16A + sizeof(_Float16)* NNODE * H_);
constexpr size_t OFF_HE   = alignup(OFF_H16B + sizeof(_Float16)* NNODE * H_);    // f16 [NEDGE*H]
constexpr size_t OFF_WE16 = alignup(OFF_HE   + sizeof(_Float16)* (size_t)NEDGE * H_);   // f16 packed [2*128*32]
constexpr size_t OFF_W116 = alignup(OFF_WE16 + sizeof(_Float16)* EPAD * H_);     // f16 packed [3*12*128*32]
constexpr size_t OFF_W216 = alignup(OFF_W116 + sizeof(_Float16)* DEPTH_ * 3 * H_ * H_);
constexpr size_t OFF_W316 = alignup(OFF_W216 + sizeof(_Float16)* DEPTH_ * H_ * H_);

// ---------------- small math helpers ----------------
struct F3 { float x, y, z; };
DEV F3 f3sub(F3 a, F3 b) { return {a.x - b.x, a.y - b.y, a.z - b.z}; }
DEV F3 f3scale(F3 a, float s) { return {a.x * s, a.y * s, a.z * s}; }
DEV float f3dot(F3 a, F3 b) { return a.x * b.x + a.y * b.y + a.z * b.z; }
DEV F3 f3cross(F3 a, F3 b) {
    return {a.y * b.z - a.z * b.y, a.z * b.x - a.x * b.z, a.x * b.y - a.y * b.x};
}
DEV F3 f3norm(F3 a) {
    float d = fmaxf(f3dot(a, a), 1e-24f);
    return f3scale(a, rsqrtf(d));
}
DEV float safesqrt(float x) { return (x > 0.f) ? sqrtf(x) : 0.f; }
DEV float fsign(float x) { return (float)((x > 0.f) - (x < 0.f)); }

DEV F3 loadXc(const float* X, int b, int n) {
    const float* p = X + ((size_t)(b * N_ + n) * 4 + 1) * 3;
    return {p[0], p[1], p[2]};
}
DEV F3 loadXd(const float* X, int b, int q) {
    int node = q / 3, atom = q % 3;
    const float* p = X + ((size_t)(b * N_ + node) * 4 + atom) * 3;
    return {p[0], p[1], p[2]};
}
DEV F3 Uvec(const float* X, const float* mask, int b, int t) {      // t in [0, N-2]
    F3 v = f3sub(loadXc(X, b, t + 1), loadXc(X, b, t));
    return f3norm(f3scale(v, mask[b * N_ + t + 1]));
}
DEV F3 Udvec(const float* X, const float* mask, int b, int t) {     // t in [0, 3N-2]
    F3 v = f3sub(loadXd(X, b, t + 1), loadXd(X, b, t));
    return f3norm(f3scale(v, mask[b * N_ + (t + 1) / 3]));
}
DEV float dihed(const float* X, const float* mask, int b, int s) {  // s in [1, 3N-3]
    F3 u2 = Udvec(X, mask, b, s - 1);
    F3 u1 = Udvec(X, mask, b, s);
    F3 u0 = Udvec(X, mask, b, s + 1);
    F3 n2 = f3norm(f3cross(u2, u1));
    F3 n1 = f3norm(f3cross(u1, u0));
    float c = f3dot(n2, n1);
    c = fminf(fmaxf(c, -1.f + 1e-7f), 1.f - 1e-7f);
    return fsign(f3dot(u2, n1)) * acosf(c);
}

// ---------------- async / wait helpers ----------------
DEV void wait_async0() {
#if defined(__has_builtin) && __has_builtin(__builtin_amdgcn_s_wait_asynccnt)
    __builtin_amdgcn_s_wait_asynccnt(0);
#else
    asm volatile("s_wait_asynccnt 0" ::: "memory");
#endif
}
DEV void async_copy_b128(unsigned lds_off, const void* gptr) {
    unsigned long long ga = (unsigned long long)(size_t)gptr;
    asm volatile("global_load_async_to_lds_b128 %0, %1, off"
                 :: "v"(lds_off), "v"(ga) : "memory");
}

// ---------------- WMMA fragment helpers (contiguous k order: lane half h holds k = 16h..16h+15, A and B alike) ----------------
DEV v16h loadA_frag(const _Float16* base, int ld, int kbase, int lane) {
    // 16-bit A 16x32: lanes 0-15 rows, K 0-15; lanes 16-31 rows, K 16-31
    int row  = lane & 15;
    int koff = kbase + ((lane & 16) ? 16 : 0);
    const _Float16* p = base + (size_t)row * ld + koff;
    v16h a;
    #pragma unroll
    for (int t = 0; t < 16; t++) a[t] = p[t];
    return a;
}
// B weights are pre-packed fragment-major: element (ks,c,kh,t) at (((ks*128+c)*2+kh)*16+t)
// representing source element row = ks*32 + kh*16 + t, col = c  (row-major K x 128).
DEV v16h loadB_packed(const _Float16* Bp, int ks, int colg, int lane) {
    const _Float16* p = Bp + ((((size_t)ks * 128 + colg) << 1) + ((lane >> 4) & 1)) * 16;
    v16h b;
    #pragma unroll
    for (int t = 0; t < 16; t++) b[t] = p[t];
    return b;
}
DEV v8f wmma16(v16h a, v16h b, v8f c) {
    return __builtin_amdgcn_wmma_f32_16x16x32_f16(false, a, false, b, (short)0, c, false, false);
}

// ---------------- kernel 1: per-node frames O and dihedral features V ----------------
__global__ void k_geom(const float* __restrict__ X, const float* __restrict__ mask,
                       float* __restrict__ O, float* __restrict__ Vf) {
    int idx = blockIdx.x * blockDim.x + threadIdx.x;
    if (idx >= NNODE) return;
    int b = idx / N_, n = idx % N_;
    float m_n = mask[idx];

    float o9[9];
    if (n >= 1 && n <= N_ - 3) {
        F3 u2 = Uvec(X, mask, b, n - 1);
        F3 u1 = Uvec(X, mask, b, n);
        F3 n2 = f3norm(f3cross(u2, u1));
        F3 o1 = f3norm(f3sub(u2, u1));
        F3 cc = f3cross(o1, n2);
        o9[0] = o1.x; o9[1] = o1.y; o9[2] = o1.z;
        o9[3] = n2.x; o9[4] = n2.y; o9[5] = n2.z;
        o9[6] = cc.x; o9[7] = cc.y; o9[8] = cc.z;
    } else {
        #pragma unroll
        for (int t = 0; t < 9; t++) o9[t] = 0.f;
    }
    #pragma unroll
    for (int t = 0; t < 9; t++) O[(size_t)idx * 9 + t] = o9[t];

    #pragma unroll
    for (int c = 0; c < 3; c++) {
        int s = 3 * n + c;
        float ang = (s >= 1 && s <= 3 * N_ - 3) ? dihed(X, mask, b, s) : 0.f;
        Vf[(size_t)idx * 6 + c]     = cosf(ang) * m_n;
        Vf[(size_t)idx * 6 + 3 + c] = sinf(ang) * m_n;
    }
}

// ---------------- kernel 2: top-K selection + edge features ----------------
__global__ void k_knn(const float* __restrict__ X, const float* __restrict__ mask,
                      const float* __restrict__ O, int* __restrict__ Eidx,
                      float* __restrict__ vmaskE, _Float16* __restrict__ Ef) {
    __shared__ float Dsel[N_];
    __shared__ float rv[256];
    __shared__ int   ri[256];
    __shared__ int   selj[K_];
    __shared__ float seld[K_];

    int node = blockIdx.x;
    int b = node / N_, i = node % N_;
    int tid = threadIdx.x;
    float m_i = mask[node];
    F3 xi = loadXc(X, b, i);

    for (int j = tid; j < N_; j += 256) {
        F3 d = f3sub(loadXc(X, b, j), xi);
        float dist = sqrtf(f3dot(d, d) + 1e-6f);
        float m2 = fmaxf(m_i * mask[b * N_ + j] - (j == i ? 1.f : 0.f), 0.f);
        Dsel[j] = m2 * dist + (1.f - m2) * 10000.f;
    }
    __syncthreads();

    for (int k = 0; k < K_; k++) {
        float bv = 3.4e38f; int bj = 0x7fffffff;
        for (int j = tid; j < N_; j += 256) {
            float v = Dsel[j];
            if (v < bv || (v == bv && j < bj)) { bv = v; bj = j; }
        }
        rv[tid] = bv; ri[tid] = bj;
        __syncthreads();
        for (int s = 128; s > 0; s >>= 1) {
            if (tid < s) {
                float v2 = rv[tid + s]; int j2 = ri[tid + s];
                if (v2 < rv[tid] || (v2 == rv[tid] && j2 < ri[tid])) { rv[tid] = v2; ri[tid] = j2; }
            }
            __syncthreads();
        }
        if (tid == 0) { selj[k] = ri[0]; seld[k] = rv[0]; Dsel[ri[0]] = 3.4e38f; }
        __syncthreads();
    }

    if (tid < K_) {
        int k = tid;
        int j = selj[k];
        float Dnb = seld[k];
        Eidx[(size_t)node * K_ + k] = j;
        float mj = mask[b * N_ + j];
        vmaskE[(size_t)node * K_ + k] = mj;

        float feat[EPAD];
        #pragma unroll
        for (int t = 0; t < EPAD; t++) feat[t] = 0.f;

        // positional encodings (16)
        float d = ((float)j - (float)i) * m_i;
        if (fabsf(d) > (float)SEQN_) d = 0.f;
        float nz = (d != 0.f) ? 1.f : 0.f;
        #pragma unroll
        for (int p = 0; p < 8; p++) {
            float freq = expf(-(float)(2 * p) * 0.57564627324851f); // ln(1e4)/16
            float ang = d * freq;
            feat[p]     = cosf(ang) * nz;
            feat[8 + p] = sinf(ang) * nz;
        }
        // RBF (16)
        #pragma unroll
        for (int r = 0; r < NRBF; r++) {
            float mu = 20.f * (float)r / 15.f;
            float t = (Dnb - mu) * (1.f / 1.25f);
            feat[16 + r] = expf(-t * t);
        }
        // orientation features (7): dU (3) + quaternion (4)
        float Oi[9], Oj[9];
        #pragma unroll
        for (int t = 0; t < 9; t++) {
            Oi[t] = O[(size_t)node * 9 + t];
            Oj[t] = O[(size_t)(b * N_ + j) * 9 + t] * m_i;
        }
        F3 xj = loadXc(X, b, j);
        F3 dxn = f3scale(f3sub(f3scale(xj, m_i), xi), m_i);
        F3 du;
        du.x = Oi[0] * dxn.x + Oi[1] * dxn.y + Oi[2] * dxn.z;
        du.y = Oi[3] * dxn.x + Oi[4] * dxn.y + Oi[5] * dxn.z;
        du.z = Oi[6] * dxn.x + Oi[7] * dxn.y + Oi[8] * dxn.z;
        du = f3norm(du);

        float R[3][3];
        #pragma unroll
        for (int a = 0; a < 3; a++)
            #pragma unroll
            for (int l = 0; l < 3; l++)
                R[a][l] = Oi[0 * 3 + a] * Oj[0 * 3 + l] + Oi[1 * 3 + a] * Oj[1 * 3 + l] +
                          Oi[2 * 3 + a] * Oj[2 * 3 + l];
        float Rxx = R[0][0], Ryy = R[1][1], Rzz = R[2][2];
        float mg0 = 0.5f * safesqrt(fabsf(1.f + Rxx - Ryy - Rzz));
        float mg1 = 0.5f * safesqrt(fabsf(1.f - Rxx + Ryy - Rzz));
        float mg2 = 0.5f * safesqrt(fabsf(1.f - Rxx - Ryy + Rzz));
        float s0 = fsign(R[2][1] - R[1][2]);
        float s1 = fsign(R[0][2] - R[2][0]);
        float s2 = fsign(R[1][0] - R[0][1]);
        float qw = 0.5f * safesqrt(fmaxf(1.f + Rxx + Ryy + Rzz, 0.f));
        float q0 = s0 * mg0, q1 = s1 * mg1, q2 = s2 * mg2;
        float qn = rsqrtf(fmaxf(q0 * q0 + q1 * q1 + q2 * q2 + qw * qw, 1e-24f));
        feat[32] = du.x * m_i; feat[33] = du.y * m_i; feat[34] = du.z * m_i;
        feat[35] = q0 * qn * m_i; feat[36] = q1 * qn * m_i;
        feat[37] = q2 * qn * m_i; feat[38] = qw * qn * m_i;

        _Float16* dst = Ef + (size_t)(node * K_ + k) * EPAD;
        #pragma unroll
        for (int t = 0; t < EPAD; t++) dst[t] = (_Float16)feat[t];
    }
}

// ---------------- kernel 3: weight conversion f32 -> f16, fragment-major packing ----------------
__global__ void k_convw(const float* __restrict__ We_w, const float* __restrict__ Wl1,
                        const float* __restrict__ Wl2, const float* __restrict__ Wl3,
                        _Float16* __restrict__ Wep, _Float16* __restrict__ W1p,
                        _Float16* __restrict__ W2p, _Float16* __restrict__ W3p) {
    int idx = blockIdx.x * blockDim.x + threadIdx.x;
    // packed index p -> (t, kh, c, ks): p = ((ks*128 + c)*2 + kh)*16 + t ; src row = ks*32+kh*16+t
    if (idx < EPAD * H_) {   // We: 2 ksteps
        int p = idx;
        int t = p & 15, kh = (p >> 4) & 1, c = (p >> 5) & 127, ks = p >> 12;
        int row = ks * 32 + kh * 16 + t;
        Wep[idx] = (_Float16)((row < E_IN) ? We_w[(size_t)row * H_ + c] : 0.f);
    }
    if (idx < DEPTH_ * 3 * H_ * H_) {  // W1: 12 ksteps per layer
        int l = idx / (3 * H_ * H_), p = idx % (3 * H_ * H_);
        int t = p & 15, kh = (p >> 4) & 1, c = (p >> 5) & 127, ks = p >> 12;
        int row = ks * 32 + kh * 16 + t;
        W1p[idx] = (_Float16)Wl1[(size_t)l * 3 * H_ * H_ + (size_t)row * H_ + c];
    }
    if (idx < DEPTH_ * H_ * H_) {      // W2/W3: 4 ksteps per layer
        int l = idx / (H_ * H_), p = idx % (H_ * H_);
        int t = p & 15, kh = (p >> 4) & 1, c = (p >> 5) & 127, ks = p >> 12;
        int row = ks * 32 + kh * 16 + t;
        W2p[idx] = (_Float16)Wl2[(size_t)l * H_ * H_ + (size_t)row * H_ + c];
        W3p[idx] = (_Float16)Wl3[(size_t)l * H_ * H_ + (size_t)row * H_ + c];
    }
}

// ---------------- kernel 4: node embedding + LayerNorm ----------------
__global__ void k_vembed(const float* __restrict__ Vf, const float* __restrict__ Wv_w,
                         const float* __restrict__ Wv_b, const float* __restrict__ gv,
                         const float* __restrict__ bv, float* __restrict__ h,
                         _Float16* __restrict__ h16) {
    __shared__ float red[128];
    int node = blockIdx.x, c = threadIdx.x;
    float acc = Wv_b[c];
    #pragma unroll
    for (int f = 0; f < 6; f++) acc += Vf[(size_t)node * 6 + f] * Wv_w[f * H_ + c];

    red[c] = acc; __syncthreads();
    for (int s = 64; s > 0; s >>= 1) { if (c < s) red[c] += red[c + s]; __syncthreads(); }
    float mu = red[0] * (1.f / 128.f); __syncthreads();
    float d0 = acc - mu;
    red[c] = d0 * d0; __syncthreads();
    for (int s = 64; s > 0; s >>= 1) { if (c < s) red[c] += red[c + s]; __syncthreads(); }
    float var = red[0] * (1.f / 127.f); __syncthreads();
    float inv = 1.f / (sqrtf(var + 1e-6f) + 1e-6f);
    float y = gv[c] * d0 * inv + bv[c];
    h[(size_t)node * H_ + c] = y;
    h16[(size_t)node * H_ + c] = (_Float16)y;
}

// ---------------- kernel 5: edge embedding GEMM (WMMA) + LayerNorm ----------------
__global__ void k_eembed(const _Float16* __restrict__ Ef, const _Float16* __restrict__ Wep,
                         const float* __restrict__ We_b, const float* __restrict__ ge,
                         const float* __restrict__ be, _Float16* __restrict__ he) {
    __shared__ alignas(16) float Ct[16 * 128];
    __shared__ float mu_s[16], inv_s[16];
    int tid = threadIdx.x, wave = tid >> 5, lane = tid & 31;
    int colg = wave * 16 + (lane & 15);

    // hoist B fragments (weights) out of the tile loop
    v16h b0 = loadB_packed(Wep, 0, colg, lane);
    v16h b1 = loadB_packed(Wep, 1, colg, lane);

    const int ntile = NEDGE / 16; // 7680
    for (int t = blockIdx.x; t < ntile; t += gridDim.x) {
        const _Float16* Abase = Ef + (size_t)t * 16 * EPAD;
        v16h a0 = loadA_frag(Abase, EPAD, 0, lane);
        v16h a1 = loadA_frag(Abase, EPAD, 32, lane);
        v8f c;
        float bb = We_b[colg];
        #pragma unroll
        for (int v = 0; v < 8; v++) c[v] = bb;
        c = wmma16(a0, b0, c);
        c = wmma16(a1, b1, c);

        int row0 = (lane >> 4) * 8;
        #pragma unroll
        for (int v = 0; v < 8; v++) Ct[(row0 + v) * 128 + colg] = c[v];
        __syncthreads();

        if (tid < 16) {
            int r = tid;
            float s = 0.f;
            for (int cc = 0; cc < 128; cc++) s += Ct[r * 128 + cc];
            float mu = s * (1.f / 128.f), vs = 0.f;
            for (int cc = 0; cc < 128; cc++) { float d = Ct[r * 128 + cc] - mu; vs += d * d; }
            mu_s[r] = mu;
            inv_s[r] = 1.f / (sqrtf(vs * (1.f / 127.f) + 1e-6f) + 1e-6f);
        }
        __syncthreads();
        for (int e = tid; e < 16 * 128; e += 256) {
            int r = e >> 7, cc = e & 127;
            float y = ge[cc] * (Ct[r * 128 + cc] - mu_s[r]) * inv_s[r] + be[cc];
            he[((size_t)t * 16 + r) * H_ + cc] = (_Float16)y;
        }
        __syncthreads();
    }
}

// ---------------- kernel 6: message-passing layer (WMMA + async LDS staging) ----------------
__global__ void k_msg(const float* __restrict__ mask, const int* __restrict__ Eidx,
                      const float* __restrict__ vmaskE, const _Float16* __restrict__ he,
                      const _Float16* __restrict__ W1, const float* __restrict__ b1,
                      const _Float16* __restrict__ W2, const float* __restrict__ b2,
                      const _Float16* __restrict__ W3, const float* __restrict__ b3,
                      const float* __restrict__ gln, const float* __restrict__ bln,
                      const float* __restrict__ hin, const _Float16* __restrict__ hin16,
                      float* __restrict__ hout, _Float16* __restrict__ hout16,
                      float* __restrict__ out, int write_out) {
    __shared__ alignas(16) _Float16 Aev[KP * 384];
    __shared__ alignas(16) _Float16 M1[KP * H_];
    __shared__ alignas(16) _Float16 M2[KP * H_];
    __shared__ alignas(16) float Sout[KP * H_];
    __shared__ int nbr[KP];
    __shared__ float vm[KP];
    __shared__ float red[128];

    int node = blockIdx.x, b = node / N_;
    int tid = threadIdx.x, wave = tid >> 5, lane = tid & 31;
    float m_i = mask[node];

    if (tid == 0) { __builtin_prefetch(W1, 0, 1); __builtin_prefetch(W2, 0, 1); }
    if (tid < KP) {
        if (tid < K_) { nbr[tid] = Eidx[(size_t)node * K_ + tid]; vm[tid] = vmaskE[(size_t)node * K_ + tid]; }
        else          { nbr[tid] = 0; vm[tid] = 0.f; }
    }
    __syncthreads();

    // build h_EV = [h_i | h_nbr | h_e]  (32 x 384, f16) in LDS
#if USE_ASYNC
    // 3 segments x 30 rows x 16 chunks of 16B, all 16B-aligned -> async global->LDS DMA
    for (int t = tid; t < 3 * K_ * 16; t += 256) {
        int sr  = t >> 4;              // seg*30 + row
        int c   = (t & 15) << 3;       // half-element offset within 128
        int seg = sr / K_;
        int r   = sr - seg * K_;
        const _Float16* gsrc;
        if (seg == 0)      gsrc = hin16 + (size_t)node * H_ + c;
        else if (seg == 1) gsrc = hin16 + (size_t)(b * N_ + nbr[r]) * H_ + c;
        else               gsrc = he + ((size_t)node * K_ + r) * H_ + c;
        unsigned ldsoff = (unsigned)(size_t)(&Aev[r * 384 + seg * 128 + c]);
        async_copy_b128(ldsoff, gsrc);
    }
    // zero pad rows 30..31
    for (int e = tid; e < 2 * 384; e += 256) Aev[K_ * 384 + e] = (_Float16)0.f;
    wait_async0();
#else
    for (int e = tid; e < KP * 384; e += 256) {
        int r = e / 384, c = e % 384;
        _Float16 v = (_Float16)0.f;
        if (r < K_) {
            if (c < 128)       v = hin16[(size_t)node * H_ + c];
            else if (c < 256)  v = hin16[(size_t)(b * N_ + nbr[r]) * H_ + (c - 128)];
            else               v = he[((size_t)node * K_ + r) * H_ + (c - 256)];
        }
        Aev[e] = v;
    }
#endif
    __syncthreads();

    int colg = wave * 16 + (lane & 15);
    int row0 = (lane >> 4) * 8;
    v8f c0, c1;

    // GEMM1: (32 x 384) @ (384 x 128) + b1, ReLU
    {
        float bb = b1[colg];
        #pragma unroll
        for (int v = 0; v < 8; v++) { c0[v] = bb; c1[v] = bb; }
        #pragma unroll
        for (int ks = 0; ks < 12; ks++) {
            v16h a0 = loadA_frag(Aev, 384, ks * 32, lane);
            v16h a1 = loadA_frag(Aev + 16 * 384, 384, ks * 32, lane);
            v16h bf = loadB_packed(W1, ks, colg, lane);
            c0 = wmma16(a0, bf, c0);
            c1 = wmma16(a1, bf, c1);
        }
        #pragma unroll
        for (int v = 0; v < 8; v++) {
            M1[(row0 + v) * H_ + colg]      = (_Float16)fmaxf(c0[v], 0.f);
            M1[(16 + row0 + v) * H_ + colg] = (_Float16)fmaxf(c1[v], 0.f);
        }
    }
    __syncthreads();

    // GEMM2: (32 x 128) @ (128 x 128) + b2, ReLU
    {
        float bb = b2[colg];
        #pragma unroll
        for (int v = 0; v < 8; v++) { c0[v] = bb; c1[v] = bb; }
        #pragma unroll
        for (int ks = 0; ks < 4; ks++) {
            v16h a0 = loadA_frag(M1, H_, ks * 32, lane);
            v16h a1 = loadA_frag(M1 + 16 * H_, H_, ks * 32, lane);
            v16h bf = loadB_packed(W2, ks, colg, lane);
            c0 = wmma16(a0, bf, c0);
            c1 = wmma16(a1, bf, c1);
        }
        #pragma unroll
        for (int v = 0; v < 8; v++) {
            M2[(row0 + v) * H_ + colg]      = (_Float16)fmaxf(c0[v], 0.f);
            M2[(16 + row0 + v) * H_ + colg] = (_Float16)fmaxf(c1[v], 0.f);
        }
    }
    __syncthreads();

    // GEMM3: (32 x 128) @ (128 x 128) + b3, * vmask
    {
        float bb = b3[colg];
        #pragma unroll
        for (int v = 0; v < 8; v++) { c0[v] = bb; c1[v] = bb; }
        #pragma unroll
        for (int ks = 0; ks < 4; ks++) {
            v16h a0 = loadA_frag(M2, H_, ks * 32, lane);
            v16h a1 = loadA_frag(M2 + 16 * H_, H_, ks * 32, lane);
            v16h bf = loadB_packed(W3, ks, colg, lane);
            c0 = wmma16(a0, bf, c0);
            c1 = wmma16(a1, bf, c1);
        }
        #pragma unroll
        for (int v = 0; v < 8; v++) {
            Sout[(row0 + v) * H_ + colg]      = c0[v] * vm[row0 + v];
            Sout[(16 + row0 + v) * H_ + colg] = c1[v] * vm[16 + row0 + v];
        }
    }
    __syncthreads();

    // mean over K + residual + LayerNorm + mask
    float x = 0.f;
    if (tid < 128) {
        float s = 0.f;
        for (int k = 0; k < K_; k++) s += Sout[k * H_ + tid];
        x = hin[(size_t)node * H_ + tid] + s * (1.f / (float)K_);
    }
    if (tid < 128) red[tid] = x;
    __syncthreads();
    for (int s = 64; s > 0; s >>= 1) { if (tid < s) red[tid] += red[tid + s]; __syncthreads(); }
    float mu = red[0] * (1.f / 128.f); __syncthreads();
    float d0 = x - mu;
    if (tid < 128) red[tid] = d0 * d0;
    __syncthreads();
    for (int s = 64; s > 0; s >>= 1) { if (tid < s) red[tid] += red[tid + s]; __syncthreads(); }
    float var = red[0] * (1.f / 127.f); __syncthreads();
    if (tid < 128) {
        float inv = 1.f / (sqrtf(var + 1e-6f) + 1e-6f);
        float y = (gln[tid] * d0 * inv + bln[tid]) * m_i;
        hout[(size_t)node * H_ + tid] = y;
        hout16[(size_t)node * H_ + tid] = (_Float16)y;
        if (write_out) out[(size_t)node * H_ + tid] = y;
    }
}

// ---------------- host launcher ----------------
extern "C" void kernel_launch(void* const* d_in, const int* in_sizes, int n_in,
                              void* d_out, int out_size, void* d_ws, size_t ws_size,
                              hipStream_t stream) {
    (void)in_sizes; (void)n_in; (void)out_size; (void)ws_size;
    const float* X    = (const float*)d_in[0];
    const float* mask = (const float*)d_in[1];
    const float* Wv_w = (const float*)d_in[2];
    const float* Wv_b = (const float*)d_in[3];
    const float* gv   = (const float*)d_in[4];
    const float* bv   = (const float*)d_in[5];
    const float* We_w = (const float*)d_in[6];
    const float* We_b = (const float*)d_in[7];
    const float* ge   = (const float*)d_in[8];
    const float* be   = (const float*)d_in[9];
    const float* Wl1  = (const float*)d_in[10];
    const float* bl1  = (const float*)d_in[11];
    const float* Wl2  = (const float*)d_in[12];
    const float* bl2  = (const float*)d_in[13];
    const float* Wl3  = (const float*)d_in[14];
    const float* bl3  = (const float*)d_in[15];
    const float* gl   = (const float*)d_in[16];
    const float* bl   = (const float*)d_in[17];

    char* w = (char*)d_ws;
    float*     O     = (float*)(w + OFF_O);
    float*     Vf    = (float*)(w + OFF_V);
    int*       Ei    = (int*)(w + OFF_EIDX);
    float*     vm    = (float*)(w + OFF_VM);
    _Float16*  Ef    = (_Float16*)(w + OFF_EF);
    float*     hA    = (float*)(w + OFF_HA);
    float*     hB    = (float*)(w + OFF_HB);
    _Float16*  h16A  = (_Float16*)(w + OFF_H16A);
    _Float16*  h16B  = (_Float16*)(w + OFF_H16B);
    _Float16*  he    = (_Float16*)(w + OFF_HE);
    _Float16*  Wep   = (_Float16*)(w + OFF_WE16);
    _Float16*  W1p   = (_Float16*)(w + OFF_W116);
    _Float16*  W2p   = (_Float16*)(w + OFF_W216);
    _Float16*  W3p   = (_Float16*)(w + OFF_W316);

    k_geom<<<(NNODE + 255) / 256, 256, 0, stream>>>(X, mask, O, Vf);
    k_convw<<<(DEPTH_ * 3 * H_ * H_ + 255) / 256, 256, 0, stream>>>(We_w, Wl1, Wl2, Wl3,
                                                                    Wep, W1p, W2p, W3p);
    k_knn<<<NNODE, 256, 0, stream>>>(X, mask, O, Ei, vm, Ef);
    k_vembed<<<NNODE, 128, 0, stream>>>(Vf, Wv_w, Wv_b, gv, bv, hA, h16A);
    k_eembed<<<960, 256, 0, stream>>>(Ef, Wep, We_b, ge, be, he);

    float* hi = hA; _Float16* hi16 = h16A;
    float* ho = hB; _Float16* ho16 = h16B;
    for (int l = 0; l < DEPTH_; l++) {
        k_msg<<<NNODE, 256, 0, stream>>>(mask, Ei, vm, he,
                                         W1p + (size_t)l * 3 * H_ * H_, bl1 + l * H_,
                                         W2p + (size_t)l * H_ * H_,     bl2 + l * H_,
                                         W3p + (size_t)l * H_ * H_,     bl3 + l * H_,
                                         gl + l * H_, bl + l * H_,
                                         hi, hi16, ho, ho16,
                                         (float*)d_out, (l == DEPTH_ - 1) ? 1 : 0);
        float* tf = hi; hi = ho; ho = tf;
        _Float16* th = hi16; hi16 = ho16; ho16 = th;
    }
}
